// LocalBlock_58102317580329
// MI455X (gfx1250) — matrix operand order repaired and checked by reading, not compiled, not run on hardware
//
#include <hip/hip_runtime.h>

// LocalBlock: per-position GEMM (B=128 x KC=896 x F=128) for 506 positions,
// fused bias + BN(affine) + ReLU.  f16 WMMA with f32 accumulation.

constexpr int Bb   = 128;   // batch
constexpr int Lf   = 512;   // input length
constexpr int Cc   = 128;   // in channels
constexpr int Ff   = 128;   // out features
constexpr int KK   = 7;     // window
constexpr int Lout = Lf - KK + 1;  // 506
constexpr float BN_EPS = 0.001f;

constexpr int LDA = Cc + 8;   // 136 halves: padded row stride for X slab  [b][c]
constexpr int LDW = Cc + 8;   // 136 halves: padded row stride for W slab  [f][c] (transposed)

typedef _Float16 v16h __attribute__((ext_vector_type(16)));
typedef _Float16 v8h  __attribute__((ext_vector_type(8)));
typedef _Float16 v4h  __attribute__((ext_vector_type(4)));
typedef float    v8f  __attribute__((ext_vector_type(8)));
typedef float    v4f  __attribute__((ext_vector_type(4)));

__global__ __launch_bounds__(256, 1)
void localblock_wmma(const float* __restrict__ x,
                     const float* __restrict__ w,
                     const float* __restrict__ bias,
                     const float* __restrict__ gamma,
                     const float* __restrict__ beta,
                     const float* __restrict__ mmean,
                     const float* __restrict__ mvar,
                     float* __restrict__ out)
{
    extern __shared__ _Float16 smem[];        // dynamic LDS: As then Ws
    _Float16* As = smem;                      // [128][LDA]  X slab, f16
    _Float16* Ws = smem + Bb * LDA;           // [128][LDW]  W slab transposed [f][c], f16

    const int l    = blockIdx.x;              // output position
    const int tid  = threadIdx.x;
    const int lane = tid & 31;
    const int wv   = tid >> 5;                // wave id 0..7
    const int g    = lane >> 4;               // lane group 0/1
    const int ln   = lane & 15;
    const int m0   = (wv & 3) * 32;           // this wave's batch-row base (2 M-tiles)
    const int n0   = (wv >> 2) * 64;          // this wave's feature base  (4 N-tiles)

    v8f acc[2][4] = {};                       // 8 16x16 f32 accumulator tiles

    for (int k = 0; k < KK; ++k) {
        // ---- stage A:  x[:, l+k, :]  ->  f16 As[b][c]  (contiguous global reads)
        {
            const int b     = tid >> 1;
            const int cbase = (tid & 1) * 64;
            const float* src = x + ((size_t)b * Lf + (l + k)) * Cc + cbase;
            _Float16*    dst = As + b * LDA + cbase;
            #pragma unroll
            for (int i = 0; i < 16; ++i) {
                v4f v = *(const v4f*)(src + i * 4);
                v4h h = { (_Float16)v.x, (_Float16)v.y, (_Float16)v.z, (_Float16)v.w };
                *(v4h*)(dst + i * 4) = h;
            }
        }
        // ---- stage W:  kernel[l, k*C + c, f] (row-major c x f) -> f16 Ws[f][c] (transposed)
        {
            const int f     = tid & 127;
            const int choff = (tid >> 7) * 4;  // 0 or 4
            const float* wsrc = w + ((size_t)l * (KK * Cc) + (size_t)k * Cc) * Ff + f;
            #pragma unroll
            for (int p = 0; p < 16; ++p) {
                const int c0 = p * 8 + choff;
                float a0 = wsrc[(size_t)(c0 + 0) * Ff];
                float a1 = wsrc[(size_t)(c0 + 1) * Ff];
                float a2 = wsrc[(size_t)(c0 + 2) * Ff];
                float a3 = wsrc[(size_t)(c0 + 3) * Ff];
                v4h h = { (_Float16)a0, (_Float16)a1, (_Float16)a2, (_Float16)a3 };
                *(v4h*)(Ws + f * LDW + c0) = h;
            }
        }
        // prefetch next chunk's weight slab into cache (global_prefetch_b8)
        if (k + 1 < KK) {
            const char* nw = (const char*)(w + ((size_t)l * (KK * Cc) + (size_t)(k + 1) * Cc) * Ff);
            __builtin_prefetch(nw + (size_t)tid * 256, 0, 1);
        }
        __syncthreads();

        // ---- compute: 4 K-steps of 32, 8 WMMAs each
        #pragma unroll
        for (int ks = 0; ks < 4; ++ks) {
            const int k0 = ks * 32;

            // A fragments (16x32 f16): lanes 0-15 hold K[0..7]+K[16..23],
            // lanes 16-31 hold K[8..15]+K[24..31]  (ISA 7.12.2)
            v16h afr[2];
            #pragma unroll
            for (int i = 0; i < 2; ++i) {
                const _Float16* ap = As + (m0 + i * 16 + ln) * LDA + k0 + g * 8;
                v8h lo = *(const v8h*)ap;
                v8h hi = *(const v8h*)(ap + 16);
                afr[i] = __builtin_shufflevector(lo, hi, 0,1,2,3,4,5,6,7,8,9,10,11,12,13,14,15);
            }
            // B fragments (32x16 f16): lane = N, lanes 0-15 -> K[0..7]+K[16..23], 16-31 -> K[8..15]+K[24..31],
            // contiguous in Ws because Ws is [f][c]
            v16h bfr[4];
            #pragma unroll
            for (int j = 0; j < 4; ++j) {
                const _Float16* bp = Ws + (n0 + j * 16 + ln) * LDW + k0 + g * 8;
                v8h lo = *(const v8h*)bp;
                v8h hi = *(const v8h*)(bp + 16);
                bfr[j] = __builtin_shufflevector(lo, hi, 0,1,2,3,4,5,6,7,8,9,10,11,12,13,14,15);
            }
            #pragma unroll
            for (int i = 0; i < 2; ++i)
                #pragma unroll
                for (int j = 0; j < 4; ++j)
                    acc[i][j] = __builtin_amdgcn_wmma_f32_16x16x32_f16(
                        false, afr[i], false, bfr[j], (short)0, acc[i][j], false, false);
        }
        __syncthreads();
    }

    // ---- epilogue: + bias, BN affine, ReLU, store
    // C/D layout: VGPR v, lane group g -> row M = g*8 + v, col N = lane&15
    #pragma unroll
    for (int j = 0; j < 4; ++j) {
        const int f   = n0 + j * 16 + ln;
        const float inv = gamma[f] * rsqrtf(mvar[f] + BN_EPS);
        const float sh  = beta[f] - mmean[f] * inv;
        const float bs  = bias[(size_t)l * Ff + f];
        #pragma unroll
        for (int i = 0; i < 2; ++i) {
            #pragma unroll
            for (int v = 0; v < 8; ++v) {
                const int row = m0 + i * 16 + g * 8 + v;   // batch index
                float y = acc[i][j][v] + bs;
                y = y * inv + sh;
                y = y > 0.0f ? y : 0.0f;
                out[((size_t)row * Lout + l) * Ff + f] = y;
            }
        }
    }
}

extern "C" void kernel_launch(void* const* d_in, const int* in_sizes, int n_in,
                              void* d_out, int out_size, void* d_ws, size_t ws_size,
                              hipStream_t stream) {
    const float* x     = (const float*)d_in[0];
    const float* kern  = (const float*)d_in[1];
    const float* bias  = (const float*)d_in[2];
    const float* gamma = (const float*)d_in[3];
    const float* beta  = (const float*)d_in[4];
    const float* mmean = (const float*)d_in[5];
    const float* mvar  = (const float*)d_in[6];
    float* out = (float*)d_out;

    const size_t lds_bytes = (size_t)(Bb * LDA + Ff * LDW) * sizeof(_Float16); // ~68 KB
    dim3 grid(Lout), block(256);
    hipLaunchKernelGGL(localblock_wmma, grid, block, lds_bytes, stream,
                       x, kern, bias, gamma, beta, mmean, mvar, out);
}
